// SwinTransformer_11261404250781
// MI455X (gfx1250) — matrix operand order repaired and checked by reading, not compiled, not run on hardware
//
#include <hip/hip_runtime.h>
#include <cstddef>

typedef _Float16 v16h __attribute__((ext_vector_type(16)));
typedef _Float16 v8h  __attribute__((ext_vector_type(8)));
typedef float    v8f  __attribute__((ext_vector_type(8)));
typedef float    v4f  __attribute__((ext_vector_type(4)));

// Build a 16-half WMMA fragment from two contiguous 16-byte LDS runs.
__device__ __forceinline__ v16h frag_ld(const _Float16* p0, const _Float16* p1) {
    v8h lo = *(const v8h*)p0;
    v8h hi = *(const v8h*)p1;
    return __builtin_shufflevector(lo, hi, 0, 1, 2, 3, 4, 5, 6, 7,
                                           8, 9, 10, 11, 12, 13, 14, 15);
}

// CDNA5 async global->LDS copy of 16 bytes per lane (ASYNCcnt-tracked).
__device__ __forceinline__ void async_b128(const _Float16* g, _Float16* lds) {
    unsigned loff = (unsigned)(uintptr_t)lds;     // low 32 bits = LDS byte offset
    asm volatile("global_load_async_to_lds_b128 %0, %1, off"
                 :: "v"(loff), "v"(g) : "memory");
}
__device__ __forceinline__ void wait_async_le4() {
    asm volatile("s_wait_asynccnt 0x4" ::: "memory");
}
__device__ __forceinline__ void wait_async_0() {
    asm volatile("s_wait_asynccnt 0x0" ::: "memory");
}

// ---------------------------------------------------------------------------
// f32 -> f16 conversion (plain, row-major)
// ---------------------------------------------------------------------------
__global__ __launch_bounds__(256) void k_f32_to_f16(const float* __restrict__ in,
                                                    _Float16* __restrict__ out, size_t n) {
    size_t i = (size_t)blockIdx.x * blockDim.x + threadIdx.x;
    if (i < n) out[i] = (_Float16)in[i];
}

// f32(K,N) -> f16 transposed (N,K): weights become row-per-output-column.
__global__ __launch_bounds__(256) void k_f32_to_f16_t(const float* __restrict__ in,
                                                      _Float16* __restrict__ out,
                                                      int K, int N) {
    size_t total = (size_t)K * N;
    size_t idx = (size_t)blockIdx.x * blockDim.x + threadIdx.x;
    if (idx >= total) return;
    int nIdx = (int)(idx % N);
    int kIdx = (int)(idx / N);
    out[(size_t)nIdx * K + kIdx] = (_Float16)in[idx];
}

// ---------------------------------------------------------------------------
// Patch embedding pack: x(B,3,224,224) f32 -> xp(B,56,56,48) f16
// ---------------------------------------------------------------------------
__global__ __launch_bounds__(256) void k_patch_pack(const float* __restrict__ x,
                                                    _Float16* __restrict__ out, int B) {
    size_t total = (size_t)B * 56 * 56 * 48;
    size_t idx = (size_t)blockIdx.x * blockDim.x + threadIdx.x;
    if (idx >= total) return;
    int cc = (int)(idx % 48); size_t r = idx / 48;
    int gx = (int)(r % 56); r /= 56;
    int gy = (int)(r % 56); int b = (int)(r / 56);
    int c = cc / 16, rem = cc % 16, py = rem / 4, px = rem % 4;
    out[idx] = (_Float16)x[(((size_t)(b * 3 + c) * 224) + gy * 4 + py) * 224 + gx * 4 + px];
}

// ---------------------------------------------------------------------------
// Patch merge gather: x(B,H,W,C) f32 -> out(B,H/2,W/2,4C) f32
// ---------------------------------------------------------------------------
__global__ __launch_bounds__(256) void k_merge_gather(const float* __restrict__ x,
                                                      float* __restrict__ out,
                                                      int B, int H, int W, int C) {
    size_t total = (size_t)B * (H / 2) * (W / 2) * 4 * C;
    size_t idx = (size_t)blockIdx.x * blockDim.x + threadIdx.x;
    if (idx >= total) return;
    int c = (int)(idx % C); size_t r = idx / C;
    int q = (int)(r % 2); r /= 2;
    int p = (int)(r % 2); r /= 2;
    int j = (int)(r % (W / 2)); r /= (W / 2);
    int i = (int)(r % (H / 2)); int b = (int)(r / (H / 2));
    out[idx] = x[(((size_t)(b * H + 2 * i + p)) * W + (2 * j + q)) * C + c];
}

// ---------------------------------------------------------------------------
// LayerNorm over last dim: one block per row; optional f32 and/or f16 output
// ---------------------------------------------------------------------------
__global__ __launch_bounds__(256) void k_layernorm(const float* __restrict__ x,
                                                   const float* __restrict__ g,
                                                   const float* __restrict__ be,
                                                   float* __restrict__ out32,
                                                   _Float16* __restrict__ out16, int C) {
    __shared__ float red[256];
    size_t row = blockIdx.x;
    const float* xr = x + row * (size_t)C;
    int tid = threadIdx.x;

    float s = 0.f;
    for (int c = tid; c < C; c += 256) s += xr[c];
    red[tid] = s; __syncthreads();
    for (int o = 128; o > 0; o >>= 1) { if (tid < o) red[tid] += red[tid + o]; __syncthreads(); }
    float mu = red[0] / (float)C;
    __syncthreads();

    float v = 0.f;
    for (int c = tid; c < C; c += 256) { float d = xr[c] - mu; v += d * d; }
    red[tid] = v; __syncthreads();
    for (int o = 128; o > 0; o >>= 1) { if (tid < o) red[tid] += red[tid + o]; __syncthreads(); }
    float rs = rsqrtf(red[0] / (float)C + 1e-5f);

    for (int c = tid; c < C; c += 256) {
        float y = (xr[c] - mu) * rs * g[c] + be[c];
        if (out32) out32[row * (size_t)C + c] = y;
        if (out16) out16[row * (size_t)C + c] = (_Float16)y;
    }
}

// ---------------------------------------------------------------------------
// WMMA GEMM: C[M,N] = act(A[M,K] @ Bt^T + bias);  Bt is f16 [N][K] (pre-transposed
// weights). Tile 64x64, BK=32, 128 thr = 4 waves (2x2, 32x32 each, 4 WMMA/step).
// Interior blocks: double-buffered LDS staged with global_load_async_to_lds_b128
// (ASYNCcnt), overlapping HBM latency with the WMMA pipe. Edge blocks: guarded
// synchronous staging. act: 0=none, 1=exact GELU.
// ---------------------------------------------------------------------------
#define GT_M 64
#define GT_N 64
#define GT_K 32

__global__ __launch_bounds__(128) void k_gemm_wmma(const _Float16* __restrict__ A,
                                                   const _Float16* __restrict__ Bt,
                                                   const float* __restrict__ bias,
                                                   float* __restrict__ c32,
                                                   _Float16* __restrict__ c16,
                                                   int M, int N, int K, int act) {
    __shared__ __align__(16) _Float16 As[2][GT_M][GT_K];   // [m][k]
    __shared__ __align__(16) _Float16 Bs[2][GT_N][GT_K];   // [n][k]

    int tid = threadIdx.x;
    int lane = tid & 31;
    int wave = tid >> 5;
    int waveM = wave >> 1, waveN = wave & 1;
    int m0 = blockIdx.y * GT_M;
    int n0 = blockIdx.x * GT_N;

    v8f acc[2][2];
    #pragma unroll
    for (int a = 0; a < 2; ++a)
        #pragma unroll
        for (int b = 0; b < 2; ++b)
            #pragma unroll
            for (int i = 0; i < 8; ++i) acc[a][b][i] = 0.0f;

    int khalf = (lane >> 4) * 8;     // A-frag K split {0-7,16-23}/{8-15,24-31}
    int kb = (lane >> 4) * 8;        // B-frag K split {0-7,16-23}/{8-15,24-31}

    bool fullTile = (m0 + GT_M <= M) && (n0 + GT_N <= N) && ((K & (GT_K - 1)) == 0);

    if (fullTile) {
        // ---------------- async double-buffered path ----------------
        int nt = K / GT_K;
        // per-thread chunk coords: 256 chunks of 8 halfs per 64x32 tile
        int rA0 = tid >> 2,          cA0 = (tid & 3) * 8;
        int rA1 = (tid + 128) >> 2,  cA1 = ((tid + 128) & 3) * 8;
        // stage tile it into buffer buf (4 async ops per thread -> 4 per wave)
        auto stage = [&](int it, int buf) {
            int k0 = it * GT_K;
            async_b128(&A [(size_t)(m0 + rA0) * K + k0 + cA0], &As[buf][rA0][cA0]);
            async_b128(&A [(size_t)(m0 + rA1) * K + k0 + cA1], &As[buf][rA1][cA1]);
            async_b128(&Bt[(size_t)(n0 + rA0) * K + k0 + cA0], &Bs[buf][rA0][cA0]);
            async_b128(&Bt[(size_t)(n0 + rA1) * K + k0 + cA1], &Bs[buf][rA1][cA1]);
        };
        stage(0, 0);
        for (int it = 0; it < nt; ++it) {
            int cur = it & 1;
            if (it + 1 < nt) { stage(it + 1, cur ^ 1); wait_async_le4(); }
            else             { wait_async_0(); }
            __syncthreads();                     // all waves' tile-it chunks visible

            v16h af[2], bf[2];
            #pragma unroll
            for (int tm = 0; tm < 2; ++tm) {
                int row = waveM * 32 + tm * 16 + (lane & 15);
                af[tm] = frag_ld(&As[cur][row][khalf], &As[cur][row][16 + khalf]);
            }
            #pragma unroll
            for (int tn = 0; tn < 2; ++tn) {
                int n = waveN * 32 + tn * 16 + (lane & 15);
                bf[tn] = frag_ld(&Bs[cur][n][kb], &Bs[cur][n][kb + 16]);
            }
            #pragma unroll
            for (int tm = 0; tm < 2; ++tm)
                #pragma unroll
                for (int tn = 0; tn < 2; ++tn)
                    acc[tm][tn] = __builtin_amdgcn_wmma_f32_16x16x32_f16(
                        false, af[tm], false, bf[tn], (short)0, acc[tm][tn], false, false);
            __syncthreads();                     // done reading buf[cur]
        }
    } else {
        // ---------------- guarded synchronous path (edges) ----------------
        for (int k0 = 0; k0 < K; k0 += GT_K) {
            for (int i = tid; i < GT_M * GT_K; i += 128) {
                int r = i >> 5, c = i & 31;
                int gm = m0 + r, gk = k0 + c;
                As[0][r][c] = (gm < M && gk < K) ? A[(size_t)gm * K + gk] : (_Float16)0.0f;
            }
            for (int i = tid; i < GT_N * GT_K; i += 128) {
                int r = i >> 5, c = i & 31;
                int gn = n0 + r, gk = k0 + c;
                Bs[0][r][c] = (gn < N && gk < K) ? Bt[(size_t)gn * K + gk] : (_Float16)0.0f;
            }
            __syncthreads();

            v16h af[2], bf[2];
            #pragma unroll
            for (int tm = 0; tm < 2; ++tm) {
                int row = waveM * 32 + tm * 16 + (lane & 15);
                af[tm] = frag_ld(&As[0][row][khalf], &As[0][row][16 + khalf]);
            }
            #pragma unroll
            for (int tn = 0; tn < 2; ++tn) {
                int n = waveN * 32 + tn * 16 + (lane & 15);
                bf[tn] = frag_ld(&Bs[0][n][kb], &Bs[0][n][kb + 16]);
            }
            #pragma unroll
            for (int tm = 0; tm < 2; ++tm)
                #pragma unroll
                for (int tn = 0; tn < 2; ++tn)
                    acc[tm][tn] = __builtin_amdgcn_wmma_f32_16x16x32_f16(
                        false, af[tm], false, bf[tn], (short)0, acc[tm][tn], false, false);
            __syncthreads();
        }
    }

    // epilogue: C/D layout -> row = base + r + 8*(lane>=16), col = lane&15
    #pragma unroll
    for (int tm = 0; tm < 2; ++tm) {
        int rowBase = m0 + waveM * 32 + tm * 16 + ((lane >> 4) * 8);
        #pragma unroll
        for (int tn = 0; tn < 2; ++tn) {
            int col = n0 + waveN * 32 + tn * 16 + (lane & 15);
            if (col >= N) continue;
            float bv = bias ? bias[col] : 0.0f;
            #pragma unroll
            for (int r = 0; r < 8; ++r) {
                int row = rowBase + r;
                if (row >= M) continue;
                float v = acc[tm][tn][r] + bv;
                if (act == 1) v = 0.5f * v * (1.0f + erff(v * 0.70710678118f));
                size_t off = (size_t)row * N + col;
                if (c32) c32[off] = v;
                if (c16) c16[off] = (_Float16)v;
            }
        }
    }
}

// ---------------------------------------------------------------------------
// Window attention: one workgroup per (batch, window, head); head_dim = 32.
// ---------------------------------------------------------------------------
__device__ __forceinline__ int swin_region(int v, int size) {
    return (v < size - 7) ? 0 : ((v < size - 3) ? 1 : 2);
}

__global__ __launch_bounds__(128) void k_window_attn(const float* __restrict__ qkv,
                                                     const float* __restrict__ pe,
                                                     _Float16* __restrict__ out16,
                                                     int size, int C, int heads, int shiftFlag) {
    __shared__ __align__(16) _Float16 Qs[64][32];    // [l][d]
    __shared__ __align__(16) _Float16 Ks[64][32];    // [l][d]  (B-frag reads along d)
    __shared__ __align__(16) _Float16 Vst[32][64];   // transposed [d][l]
    __shared__ __align__(16) float    Ss[64][64];
    __shared__ __align__(16) _Float16 Ps[64][64];

    int n = size / 7;
    int nWin = n * n;
    int id = blockIdx.x;
    int head = id % heads; id /= heads;
    int win = id % nWin;   id /= nWin;
    int b = id;
    int wy = win / n, wx = win % n;
    int sh = shiftFlag ? 3 : 0;

    int tid = threadIdx.x;
    int lane = tid & 31;
    int wave = tid >> 5;

    // load q,k,v with float4; softmax scale folded into q
    for (int i = tid; i < 64 * 8; i += 128) {
        int l = i >> 3, d4 = (i & 7) * 4;
        v4f qv, kv, vv;
        #pragma unroll
        for (int u = 0; u < 4; ++u) { qv[u] = 0.f; kv[u] = 0.f; vv[u] = 0.f; }
        if (l < 49) {
            int ly = l / 7, lx = l % 7;
            int h = (wy * 7 + ly - sh + size) % size;
            int w = (wx * 7 + lx - sh + size) % size;
            size_t t = ((size_t)(b * size + h)) * size + w;
            const float* base = qkv + t * (size_t)(3 * C) + head * 32 + d4;
            qv = *(const v4f*)(base);
            kv = *(const v4f*)(base + C);
            vv = *(const v4f*)(base + 2 * C);
        }
        #pragma unroll
        for (int u = 0; u < 4; ++u) {
            Qs[l][d4 + u] = (_Float16)(qv[u] * 0.17677669529f);  // 1/sqrt(32)
            Ks[l][d4 + u] = (_Float16)kv[u];
            Vst[d4 + u][l] = (_Float16)vv[u];
        }
    }
    __syncthreads();

    // ---- scores: S = (q*scale) @ k^T  (wave strip = 16 rows x 64 cols) ----
    v8f accS[4];
    #pragma unroll
    for (int t = 0; t < 4; ++t)
        #pragma unroll
        for (int i = 0; i < 8; ++i) accS[t][i] = 0.0f;

    {
        int row = wave * 16 + (lane & 15);
        int khalf = (lane >> 4) * 8;
        int kb = (lane >> 4) * 8;
        v16h aq = frag_ld(&Qs[row][khalf], &Qs[row][16 + khalf]);
        #pragma unroll
        for (int t = 0; t < 4; ++t) {
            int nn = t * 16 + (lane & 15);
            v16h bk = frag_ld(&Ks[nn][kb], &Ks[nn][kb + 16]);   // B[k][n] = K[n][k]
            accS[t] = __builtin_amdgcn_wmma_f32_16x16x32_f16(
                false, aq, false, bk, (short)0, accS[t], false, false);
        }
    }

    // write S + relative-PE bias (+ shift mask); pad cols -> -1e30
    {
        int iBase = wave * 16 + ((lane >> 4) * 8);
        #pragma unroll
        for (int t = 0; t < 4; ++t) {
            int j0 = t * 16 + (lane & 15);
            #pragma unroll
            for (int r = 0; r < 8; ++r) {
                int i = iBase + r;
                float v;
                if (i < 49 && j0 < 49) {
                    int yi = i / 7, xi = i % 7, yj = j0 / 7, xj = j0 % 7;
                    int pidx = (yi - yj + 6) * 13 + (xi - xj + 6);
                    v = accS[t][r] + pe[head * 169 + pidx];
                    if (shiftFlag) {
                        int ci = swin_region(wy * 7 + yi, size) * 3 + swin_region(wx * 7 + xi, size);
                        int cj = swin_region(wy * 7 + yj, size) * 3 + swin_region(wx * 7 + xj, size);
                        if (ci != cj) v += -100.0f;
                    }
                } else {
                    v = -1e30f;
                }
                Ss[i][j0] = v;
            }
        }
    }
    __syncthreads();

    // ---- softmax over 64 cols (padded cols vanish) ----
    if (tid < 64) {
        int i = tid;
        float mx = -1e30f;
        for (int j = 0; j < 64; ++j) mx = fmaxf(mx, Ss[i][j]);
        float sum = 0.f;
        for (int j = 0; j < 64; ++j) { float e = __expf(Ss[i][j] - mx); Ss[i][j] = e; sum += e; }
        float inv = 1.0f / sum;
        for (int j = 0; j < 64; ++j) Ps[i][j] = (_Float16)(Ss[i][j] * inv);
    }
    __syncthreads();

    // ---- O = P(64x64) @ V(64x32): 2 K-steps x 2 N-tiles ----
    v8f accO[2];
    #pragma unroll
    for (int t = 0; t < 2; ++t)
        #pragma unroll
        for (int i = 0; i < 8; ++i) accO[t][i] = 0.0f;

    #pragma unroll
    for (int ks = 0; ks < 2; ++ks) {
        int row = wave * 16 + (lane & 15);
        int khalf = ks * 32 + (lane >> 4) * 8;
        int kb = ks * 32 + (lane >> 4) * 8;
        v16h ap = frag_ld(&Ps[row][khalf], &Ps[row][16 + khalf]);
        #pragma unroll
        for (int t = 0; t < 2; ++t) {
            int nn = t * 16 + (lane & 15);
            v16h bv = frag_ld(&Vst[nn][kb], &Vst[nn][kb + 16]); // B[k][n] = Vst[n][k]
            accO[t] = __builtin_amdgcn_wmma_f32_16x16x32_f16(
                false, ap, false, bv, (short)0, accO[t], false, false);
        }
    }

    // store f16 output (proj GEMM input); same token mapping un-rolls shift
    {
        int iBase = wave * 16 + ((lane >> 4) * 8);
        #pragma unroll
        for (int t = 0; t < 2; ++t) {
            int d = t * 16 + (lane & 15);
            #pragma unroll
            for (int r = 0; r < 8; ++r) {
                int i = iBase + r;
                if (i < 49) {
                    int ly = i / 7, lx = i % 7;
                    int h = (wy * 7 + ly - sh + size) % size;
                    int w = (wx * 7 + lx - sh + size) % size;
                    size_t tok = ((size_t)(b * size + h)) * size + w;
                    out16[tok * (size_t)C + head * 32 + d] = (_Float16)accO[t][r];
                }
            }
        }
    }
}

// ---------------------------------------------------------------------------
// Head mean over spatial: in(B,L,C) f32 -> out(B,C)
// ---------------------------------------------------------------------------
__global__ __launch_bounds__(256) void k_head_mean(const float* __restrict__ in,
                                                   float* __restrict__ out,
                                                   int B, int L, int C) {
    int i = blockIdx.x * blockDim.x + threadIdx.x;
    if (i >= B * C) return;
    int b = i / C, c = i % C;
    float s = 0.f;
    for (int t = 0; t < L; ++t) s += in[((size_t)b * L + t) * C + c];
    out[i] = s / (float)L;
}

// ---------------------------------------------------------------------------
// Host orchestration
// ---------------------------------------------------------------------------
struct BlkP { const float *n1g,*n1b,*qkvw,*qkvb,*projw,*projb,*pe,*n2g,*n2b,*w1,*b1,*w2,*b2; };
struct MrgP { const float *ng,*nb,*w; };

extern "C" void kernel_launch(void* const* d_in, const int* in_sizes, int n_in,
                              void* d_out, int out_size, void* d_ws, size_t ws_size,
                              hipStream_t stream) {
    (void)n_in; (void)out_size; (void)ws_size;
    const int depths[4] = {2, 2, 6, 2};

    const float* x = nullptr;
    const float *patch_w, *patch_b, *eng, *enb, *hng, *hnb;
    BlkP blk[12]; MrgP mrg[3];

    int cur = 0;
    auto nxt = [&]() { return (const float*)d_in[cur++]; };

    if (in_sizes[0] == 32 * 3 * 224 * 224) {
        // insertion-order flattening
        x = nxt();
        patch_w = nxt(); patch_b = nxt();
        eng = nxt(); enb = nxt();
        hng = nxt(); hnb = nxt();
        int k = 0;
        for (int st = 0; st < 4; ++st) {
            if (st > 0) { mrg[st-1].ng = nxt(); mrg[st-1].nb = nxt(); mrg[st-1].w = nxt(); }
            for (int j = 0; j < depths[st]; ++j, ++k) {
                BlkP& b = blk[k];
                b.n1g = nxt(); b.n1b = nxt(); b.qkvw = nxt(); b.qkvb = nxt();
                b.projw = nxt(); b.projb = nxt(); b.pe = nxt();
                b.n2g = nxt(); b.n2b = nxt();
                b.w1 = nxt(); b.b1 = nxt(); b.w2 = nxt(); b.b2 = nxt();
            }
        }
    } else {
        // JAX pytree (sorted-key) flattening: params first, 'x' last
        enb = nxt(); eng = nxt(); hnb = nxt(); hng = nxt();
        patch_b = nxt(); patch_w = nxt();
        int k = 0;
        for (int st = 0; st < 4; ++st) {
            for (int j = 0; j < depths[st]; ++j, ++k) {
                BlkP& b = blk[k];
                b.b1 = nxt(); b.b2 = nxt(); b.w1 = nxt(); b.w2 = nxt();
                b.n1b = nxt(); b.n1g = nxt(); b.n2b = nxt(); b.n2g = nxt();
                b.pe = nxt(); b.projb = nxt(); b.projw = nxt();
                b.qkvb = nxt(); b.qkvw = nxt();
            }
            if (st > 0) { mrg[st-1].nb = nxt(); mrg[st-1].ng = nxt(); mrg[st-1].w = nxt(); }
        }
        x = nxt();
    }

    // workspace carve
    size_t off = 0;
    auto walloc = [&](size_t bytes) -> void* {
        void* p = (void*)((char*)d_ws + off);
        off += (bytes + 255) & ~(size_t)255;
        return p;
    };
    // weight conversion: f32 (K,N) -> f16 transposed (N,K)
    auto cvtT = [&](const float* src, int K, int N) -> _Float16* {
        _Float16* dst = (_Float16*)walloc((size_t)K * N * sizeof(_Float16));
        size_t tot = (size_t)K * N;
        k_f32_to_f16_t<<<(unsigned)((tot + 255) / 256), 256, 0, stream>>>(src, dst, K, N);
        return dst;
    };

    // f16 transposed weights
    _Float16* patchw16 = cvtT(patch_w, 48, 96);
    _Float16 *qkvw16[12], *projw16[12], *w116[12], *w216[12], *mergew16[3];
    {
        int C = 96, k = 0;
        for (int st = 0; st < 4; ++st) {
            if (st > 0) { mergew16[st-1] = cvtT(mrg[st-1].w, 4 * C, 2 * C); C *= 2; }
            for (int j = 0; j < depths[st]; ++j, ++k) {
                qkvw16[k]  = cvtT(blk[k].qkvw,  C, 3 * C);
                projw16[k] = cvtT(blk[k].projw, C, C);
                w116[k]    = cvtT(blk[k].w1,    C, 4 * C);
                w216[k]    = cvtT(blk[k].w2,    4 * C, C);
            }
        }
    }

    // activation regions (sized for stage 0: T0 tokens, C0=96)
    const size_t T0 = (size_t)32 * 56 * 56;
    float*    R0 = (float*)   walloc(T0 * 96 * sizeof(float));      // current x (f32)
    _Float16* R1 = (_Float16*)walloc(T0 * 96 * sizeof(_Float16));   // LN outputs (f16)
    float*    R2 = (float*)   walloc(T0 * 288 * sizeof(float));     // qkv / mlp-hidden
    _Float16* R3 = (_Float16*)walloc(T0 * 96 * sizeof(_Float16));   // attn out / patch pack

    auto gemm = [&](const _Float16* A, const _Float16* Bt, const float* bias,
                    float* c32, _Float16* c16, int M, int N, int K, int act) {
        dim3 g((N + GT_N - 1) / GT_N, (M + GT_M - 1) / GT_M);
        k_gemm_wmma<<<g, 128, 0, stream>>>(A, Bt, bias, c32, c16, M, N, K, act);
    };
    auto ln = [&](const float* xx, const float* g_, const float* b_,
                  float* o32, _Float16* o16, size_t T, int C) {
        k_layernorm<<<(unsigned)T, 256, 0, stream>>>(xx, g_, b_, o32, o16, C);
    };

    // ---- patch embed ----
    {
        size_t tot = T0 * 48;
        k_patch_pack<<<(unsigned)((tot + 255) / 256), 256, 0, stream>>>(x, R3, 32);
        gemm(R3, patchw16, patch_b, R2, nullptr, (int)T0, 96, 48, 0);
        ln(R2, eng, enb, R0, nullptr, T0, 96);
    }

    // ---- stages ----
    size_t T = T0; int C = 96, sz = 56, hd = 3, bi = 0;
    for (int st = 0; st < 4; ++st) {
        if (st > 0) {
            size_t tot = T * C;  // == (T/4) * 4C
            k_merge_gather<<<(unsigned)((tot + 255) / 256), 256, 0, stream>>>(R0, R2, 32, sz, sz, C);
            ln(R2, mrg[st-1].ng, mrg[st-1].nb, nullptr, R1, T / 4, 4 * C);
            gemm(R1, mergew16[st-1], nullptr, R0, nullptr, (int)(T / 4), 2 * C, 4 * C, 0);
            T /= 4; C *= 2; sz /= 2; hd *= 2;
        }
        int nWin = (sz / 7) * (sz / 7);
        for (int j = 0; j < depths[st]; ++j, ++bi) {
            const BlkP& b = blk[bi];
            ln(R0, b.n1g, b.n1b, nullptr, R1, T, C);
            gemm(R1, qkvw16[bi], b.qkvb, R2, nullptr, (int)T, 3 * C, C, 0);
            k_window_attn<<<32 * nWin * hd, 128, 0, stream>>>(R2, b.pe, R3, sz, C, hd, (j % 2) == 1);
            gemm(R3, projw16[bi], b.projb, R0, nullptr, (int)T, C, C, 0);
            ln(R0, b.n2g, b.n2b, nullptr, R1, T, C);
            gemm(R1, w116[bi], b.b1, nullptr, (_Float16*)R2, (int)T, 4 * C, C, 1);  // GELU
            gemm((_Float16*)R2, w216[bi], b.b2, R0, nullptr, (int)T, C, 4 * C, 0);
        }
    }

    // ---- head: LN + spatial mean ----
    ln(R0, hng, hnb, R2, nullptr, T, C);
    k_head_mean<<<(32 * 768 + 255) / 256, 256, 0, stream>>>(R2, (float*)d_out, 32, sz * sz, C);
}
